// FreqCA_38946763441015
// MI455X (gfx1250) — matrix operand order repaired and checked by reading, not compiled, not run on hardware
//
#include <hip/hip_runtime.h>
#include <hip/hip_bf16.h>
#include <math.h>
#include <stdint.h>

#define TPB 256

typedef __attribute__((ext_vector_type(16))) __bf16 v16bf;
typedef __attribute__((ext_vector_type(8)))  __bf16 v8bf;
typedef __attribute__((ext_vector_type(8)))  float  v8f;

constexpr int CB  = 8;              // batch
constexpr int CC  = 256;            // fea channels
constexpr int CH  = 128;
constexpr int CWW = 128;
constexpr int CHW = CH * CWW;       // 16384
constexpr int CQKV = 3 * CC;        // 768
constexpr int CFREQ = CH * (CWW/2 + 1);   // 128*65 = 8320
constexpr float ATT_SCALE = 0.17677669529663687f;  // 1/sqrt(32)

__device__ __forceinline__ v8f wmma_bf16(const v16bf& a, const v16bf& b, const v8f& c) {
  return __builtin_amdgcn_wmma_f32_16x16x32_bf16(false, a, false, b, (short)0, c,
                                                 false, false);
}

// ---------------------------------------------------------------------------
// fp32 -> bf16 weight conversion (1x1 weights: K ordering is plain ic)
// ---------------------------------------------------------------------------
__global__ void k_cvt_bf16(const float* __restrict__ s, __bf16* __restrict__ d, int n) {
  int i = blockIdx.x * TPB + threadIdx.x;
  if (i < n) d[i] = (__bf16)s[i];
}

// 3x3 weights: convert AND permute K to tap-major: dst k = tap*Cin + ic.
// src layout [oc][ic][3][3]  ->  dst [oc][tap][ic]
__global__ void k_cvt_w3(const float* __restrict__ s, __bf16* __restrict__ d,
                         int Cout, int Cin) {
  int i = blockIdx.x * TPB + threadIdx.x;
  int tot = Cout * Cin * 9;
  if (i >= tot) return;
  int oc  = i / (Cin * 9);
  int rem = i - oc * Cin * 9;
  int ic  = rem / 9, r = rem - ic * 9;
  d[(size_t)oc * Cin * 9 + (size_t)r * Cin + ic] = (__bf16)s[i];
}

// ---------------------------------------------------------------------------
// GroupNorm: one block per (b, group); 32 groups of 8 channels.
// ---------------------------------------------------------------------------
__global__ void k_groupnorm(const float* __restrict__ x, const float* __restrict__ gw,
                            const float* __restrict__ gb, float* __restrict__ y) {
  __shared__ float s1[TPB], s2[TPB];
  const int b = blockIdx.x >> 5, g = blockIdx.x & 31;
  const size_t base = ((size_t)b * CC + g * 8) * CHW;
  const int NG = 8 * CHW;
  float a0 = 0.f, a1 = 0.f;
  for (int i = threadIdx.x; i < NG; i += TPB) {
    float v = x[base + i]; a0 += v; a1 += v * v;
  }
  s1[threadIdx.x] = a0; s2[threadIdx.x] = a1; __syncthreads();
  for (int off = TPB / 2; off > 0; off >>= 1) {
    if (threadIdx.x < off) {
      s1[threadIdx.x] += s1[threadIdx.x + off];
      s2[threadIdx.x] += s2[threadIdx.x + off];
    }
    __syncthreads();
  }
  const float mean = s1[0] / (float)NG;
  const float var  = s2[0] / (float)NG - mean * mean;
  const float inv  = rsqrtf(var + 1e-5f);
  for (int i = threadIdx.x; i < NG; i += TPB) {
    int c = g * 8 + (i >> 14);
    y[base + i] = (x[base + i] - mean) * inv * gw[c] + gb[c];
  }
}

// ---------------------------------------------------------------------------
// Depthwise 3x3, pad 1, no bias.  Output bf16 (feeds the QKV WMMA GEMM only).
// ---------------------------------------------------------------------------
__global__ void k_dwconv3(const float* __restrict__ x, const float* __restrict__ w,
                          __bf16* __restrict__ y) {
  int idx = blockIdx.x * TPB + threadIdx.x;          // [0, 8*256*16384)
  int pix = idx & 16383;
  int px  = pix & 127, py = pix >> 7;
  int c   = (idx >> 14) & 255;
  size_t pb = (size_t)(idx >> 14) * CHW;             // (b*256+c)*CHW
  const float* wc = w + c * 9;
  float acc = 0.f;
#pragma unroll
  for (int ky = 0; ky < 3; ++ky)
#pragma unroll
    for (int kx = 0; kx < 3; ++kx) {
      int yy = py + ky - 1, xx = px + kx - 1;
      if ((unsigned)yy < 128u && (unsigned)xx < 128u)
        acc += wc[ky * 3 + kx] * x[pb + (yy << 7) + xx];
    }
  y[idx] = (__bf16)acc;
}

// ---------------------------------------------------------------------------
// Unified WMMA conv kernel with software-pipelined double buffering.
//   - ntaps==1: 1x1 conv (plain GEMM, K=Cin)
//   - ntaps==9: 3x3 conv pad 1; tap-major weights => each 64-wide k-step has
//               one fixed (dy,dx) == shifted 1x1 GEMM step.
// Block: 8 waves, 32(M) x 128(N = one image row); per 64-K round:
//   B^T tile [128n][64k] staged transposed in LDS (aligned b128 global loads,
//   tap shift via LDS store index, borders zero-filled);
//   A tile [32m][64k] staged via async global->LDS DMA (ASYNCcnt path).
// Ping-pong buffers: stage(step+1) overlaps the 4 WMMAs of step; a single
// barrier per round; per-wave s_wait_asynccnt deferred to next round's top.
// ---------------------------------------------------------------------------
__global__ void k_conv_wmma(const __bf16* __restrict__ Wt, const __bf16* __restrict__ X,
                            const float* __restrict__ bias, const float* __restrict__ res,
                            float* __restrict__ out32, __bf16* __restrict__ out16,
                            int M, int Cin, int ntaps, int act) {
  constexpr int N = CHW;
  constexpr int LDT = 72;                     // padded LDS row stride (bf16)
  __shared__ __align__(16) __bf16 sBT[2][128 * LDT];  // B^T ping-pong  36KB
  __shared__ __align__(16) __bf16 sAT[2][32 * LDT];   // A   ping-pong   9KB
  const int K = ntaps * Cin;
  const int z = blockIdx.z;
  const size_t xb = (size_t)z * (size_t)Cin * CHW;
  const size_t ob = (size_t)z * (size_t)M * N;
  const int tid  = threadIdx.x;
  const int lane = tid & 31;
  const int wave = tid >> 5;
  const int wm = wave >> 2, wn = wave & 3;
  const int m0b  = blockIdx.y * 32;
  const int yrow = blockIdx.x;                // one output image row per block
  const int nblk = yrow * 128;
  const int hi = lane >> 4, lm = lane & 15;
  const int skl = tid >> 3;                   // staging: k sub-row 0..31
  const int snl = (tid & 7) * 16;             // staging: source col base

  const int icShift = (Cin == 512) ? 3 : 2;   // (Cin/64) as a shift
  const int icMask  = (1 << icShift) - 1;
  const int nsteps  = ntaps << icShift;

  v8f acc0 = {}, acc1 = {};

  // ---- stage one 64-K tile (B^T via ds stores, A via async DMA) ----
  auto stage = [&](int buf, int step) {
    const int t   = step >> icShift;
    const int ic0 = (step & icMask) << 6;
    const int dy = (ntaps == 1) ? 0 : (t / 3 - 1);
    const int dx = (ntaps == 1) ? 0 : (t - (t / 3) * 3 - 1);
    const int yy = yrow + dy;
    const bool rowOk = (unsigned)yy < 128u;
    __bf16* BT = &sBT[buf][0];
#pragma unroll
    for (int h = 0; h < 2; ++h) {
      int kk = skl + h * 32;
      if (!rowOk) {
#pragma unroll
        for (int j = 0; j < 16; ++j) BT[(snl + j) * LDT + kk] = (__bf16)0.f;
      } else {
        const __bf16* src = X + xb + ((size_t)(ic0 + kk) << 14) + (yy << 7) + snl;
        v8bf c0 = *reinterpret_cast<const v8bf*>(src);
        v8bf c1 = *reinterpret_cast<const v8bf*>(src + 8);
        if (dx == 0) {
#pragma unroll
          for (int j = 0; j < 8; ++j) {
            BT[(snl + j) * LDT + kk]     = c0[j];
            BT[(snl + 8 + j) * LDT + kk] = c1[j];
          }
        } else {
#pragma unroll
          for (int j = 0; j < 16; ++j) {       // out col n = src col - dx
            int n = snl + j - dx;
            __bf16 v = (j < 8) ? c0[j] : c1[j - 8];
            if ((unsigned)n < 128u) BT[n * LDT + kk] = v;
          }
          if ((tid & 7) == 0) {                // edge col never written: zero
            int ne = (dx > 0) ? 127 : 0;
            BT[ne * LDT + kk] = (__bf16)0.f;
          }
        }
      }
    }
    {
      int row = wave * 4 + (lane >> 3);
      int cb  = (lane & 7) * 8;                // bf16 col in tile
      unsigned long long g = (unsigned long long)(uintptr_t)
          (Wt + (size_t)(m0b + row) * K + (size_t)t * Cin + ic0 + cb);
      unsigned l = (unsigned)(uintptr_t)&sAT[buf][0] +
                   (unsigned)(row * LDT + cb) * 2u;
      asm volatile("global_load_async_to_lds_b128 %0, %1, off"
                   :: "v"(l), "v"(g) : "memory");
    }
  };

  // ---- consume one staged tile: build fragments (ISA 7.12.2) + 4 WMMA ----
  auto compute = [&](int buf) {
    const __bf16* AT = &sAT[buf][0];
    const __bf16* BT = &sBT[buf][0];
    v16bf a0, a1, b00, b01, b10, b11;
    {
      const __bf16* ar = AT + (wm * 16 + lm) * LDT;
      v8bf t0 = *reinterpret_cast<const v8bf*>(ar + hi * 16);
      v8bf t1 = *reinterpret_cast<const v8bf*>(ar + hi * 16 + 8);
      v8bf t2 = *reinterpret_cast<const v8bf*>(ar + 32 + hi * 16);
      v8bf t3 = *reinterpret_cast<const v8bf*>(ar + 32 + hi * 16 + 8);
#pragma unroll
      for (int e = 0; e < 8; ++e) {
        a0[e] = t0[e]; a0[e + 8] = t1[e];
        a1[e] = t2[e]; a1[e + 8] = t3[e];
      }
    }
    {
      const __bf16* br0 = BT + (wn * 32 + lm) * LDT;
      const __bf16* br1 = BT + (wn * 32 + 16 + lm) * LDT;
      v8bf p0 = *reinterpret_cast<const v8bf*>(br0 + hi * 16);
      v8bf p1 = *reinterpret_cast<const v8bf*>(br0 + hi * 16 + 8);
      v8bf p2 = *reinterpret_cast<const v8bf*>(br0 + 32 + hi * 16);
      v8bf p3 = *reinterpret_cast<const v8bf*>(br0 + 32 + hi * 16 + 8);
      v8bf q0 = *reinterpret_cast<const v8bf*>(br1 + hi * 16);
      v8bf q1 = *reinterpret_cast<const v8bf*>(br1 + hi * 16 + 8);
      v8bf q2 = *reinterpret_cast<const v8bf*>(br1 + 32 + hi * 16);
      v8bf q3 = *reinterpret_cast<const v8bf*>(br1 + 32 + hi * 16 + 8);
#pragma unroll
      for (int e = 0; e < 8; ++e) {
        b00[e] = p0[e]; b00[e + 8] = p1[e];
        b01[e] = p2[e]; b01[e + 8] = p3[e];
        b10[e] = q0[e]; b10[e + 8] = q1[e];
        b11[e] = q2[e]; b11[e + 8] = q3[e];
      }
    }
    acc0 = wmma_bf16(a0, b00, acc0);
    acc0 = wmma_bf16(a1, b01, acc0);
    acc1 = wmma_bf16(a0, b10, acc1);
    acc1 = wmma_bf16(a1, b11, acc1);
  };

  // ---- pipelined main loop: one barrier per 64-K round ----
  stage(0, 0);
#pragma unroll 2
  for (int step = 0; step < nsteps; ++step) {
    asm volatile("s_wait_asynccnt 0x0" ::: "memory");  // our DMA from stage()
    __syncthreads();                                   // staged tile visible
    if (step + 1 < nsteps) stage((step + 1) & 1, step + 1);
    compute(step & 1);
  }

  // ---- epilogue: optional bias / residual / fp32 and bf16 outputs ----
#pragma unroll
  for (int tt = 0; tt < 2; ++tt) {
    const v8f& acc = tt ? acc1 : acc0;
    int n = nblk + wn * 32 + tt * 16 + lm;
#pragma unroll
    for (int r = 0; r < 8; ++r) {
      int m = m0b + wm * 16 + hi * 8 + r;
      float v = acc[r];
      if (bias)  v += bias[m];
      if (res)   v += res[ob + (size_t)m * N + n];
      if (act)   v = v / (1.f + expf(-v));
      if (out32) out32[ob + (size_t)m * N + n] = v;
      if (out16) out16[ob + (size_t)m * N + n] = (__bf16)v;
    }
  }
}

// ---------------------------------------------------------------------------
// L2-normalize across 256 channels of one of q/k/v inside [b,768,HW] (inplace)
// ---------------------------------------------------------------------------
__global__ void k_l2norm(float* __restrict__ t, int coff) {
  int gid = blockIdx.x * TPB + threadIdx.x;          // 131072 threads
  int b = gid >> 14, pix = gid & 16383;
  size_t base = ((size_t)b * CQKV + coff) * CHW + pix;
  float s = 0.f;
  for (int c = 0; c < CC; ++c) {
    float v = t[base + ((size_t)c << 14)]; s += v * v;
  }
  float inv = 1.f / fmaxf(sqrtf(s), 1e-12f);
  for (int c = 0; c < CC; ++c) t[base + ((size_t)c << 14)] *= inv;
}

// ---------------------------------------------------------------------------
// In-LDS 128x128 2D FFT (radix-2 DIT); one plane per workgroup, 128KB LDS.
// ---------------------------------------------------------------------------
__device__ void fft128(float* re, float* im, int lstride, int estride, float sign) {
  const int tid = threadIdx.x;
  for (int i = tid; i < 128 * 128; i += TPB) {       // per-line bit reversal
    int l = i >> 7, e = i & 127;
    int j = (int)(__brev((unsigned)e) >> 25);
    if (j > e) {
      int i1 = l * lstride + e * estride;
      int i2 = l * lstride + j * estride;
      float tr = re[i1]; re[i1] = re[i2]; re[i2] = tr;
      float ti = im[i1]; im[i1] = im[i2]; im[i2] = ti;
    }
  }
  __syncthreads();
  for (int s = 0; s < 7; ++s) {
    int half = 1 << s;
    for (int t = tid; t < 128 * 64; t += TPB) {
      int l = t >> 6, j = t & 63;
      int grp = j >> s, pos = j & (half - 1);
      int i1 = l * lstride + (grp * (half << 1) + pos) * estride;
      int i2 = i1 + half * estride;
      float ang = sign * 3.14159265358979323846f * (float)pos / (float)half;
      float c = cosf(ang), sn = sinf(ang);
      float xr = re[i2], xi = im[i2];
      float tr = xr * c - xi * sn;
      float ti = xr * sn + xi * c;
      float ur = re[i1], ui = im[i1];
      re[i1] = ur + tr; im[i1] = ui + ti;
      re[i2] = ur - tr; im[i2] = ui - ti;
    }
    __syncthreads();
  }
}

// Forward rfft2 (ortho) of one plane of qkv; emit amp/phase [128,65].
__global__ void k_rfft2_ap(const float* __restrict__ x, int coff,
                           float* __restrict__ amp, float* __restrict__ ph) {
  extern __shared__ float smem[];
  float* re = smem; float* im = smem + CHW;
  const int p = blockIdx.x;                          // 0..2047 = (b,c)
  const int b = p >> 8, c = p & 255;
  const size_t pb = ((size_t)b * CQKV + coff + c) * CHW;
  const size_t fb = (size_t)p * CFREQ;
  for (int i = threadIdx.x; i < CHW; i += TPB) { re[i] = x[pb + i]; im[i] = 0.f; }
  __syncthreads();
  fft128(re, im, 128, 1, -1.f);    // rows  (along W)
  fft128(re, im, 1, 128, -1.f);    // cols  (along H)
  for (int i = threadIdx.x; i < 128 * 65; i += TPB) {
    int y = i / 65, kx = i - y * 65;
    float r = re[(y << 7) + kx] * (1.f / 128.f);
    float q = im[(y << 7) + kx] * (1.f / 128.f);
    amp[fb + i] = sqrtf(r * r + q * q);
    ph[fb + i]  = atan2f(q, r);
  }
}

// Inverse rfft2 (ortho) from amp*exp(i*phase); bf16 output (feeds ao GEMM).
__global__ void k_irfft2(const float* __restrict__ amp, const float* __restrict__ ph,
                         __bf16* __restrict__ out) {
  extern __shared__ float smem[];
  float* re = smem; float* im = smem + CHW;
  const int p = blockIdx.x;
  const size_t pb = (size_t)p * CHW;
  const size_t fb = (size_t)p * CFREQ;
  for (int i = threadIdx.x; i < CHW; i += TPB) {
    int y = i >> 7, kx = i & 127;
    float a, pp, sgn = 1.f;
    if (kx <= 64) { a = amp[fb + y * 65 + kx]; pp = ph[fb + y * 65 + kx]; }
    else {
      int y2 = (128 - y) & 127, x2 = 128 - kx;
      a = amp[fb + y2 * 65 + x2]; pp = ph[fb + y2 * 65 + x2]; sgn = -1.f;
    }
    re[i] = a * cosf(pp);
    im[i] = sgn * a * sinf(pp);
  }
  __syncthreads();
  fft128(re, im, 128, 1, 1.f);
  fft128(re, im, 1, 128, 1.f);
  for (int i = threadIdx.x; i < CHW; i += TPB) out[pb + i] = (__bf16)(re[i] * (1.f / 128.f));
}

// ---------------------------------------------------------------------------
// Channel attention per (b,head): sim = softmax(q k^T * scale); out = sim v.
// ---------------------------------------------------------------------------
__global__ void k_attn(const float* __restrict__ q, const float* __restrict__ k,
                       const float* __restrict__ v, float* __restrict__ out) {
  __shared__ float sim[32][32];
  const size_t base = (size_t)blockIdx.x * 32 * CFREQ;   // (b*8+h)*32 channels
  for (int e = threadIdx.x; e < 1024; e += TPB) {
    int i = e >> 5, j = e & 31;
    const float* qi = q + base + (size_t)i * CFREQ;
    const float* kj = k + base + (size_t)j * CFREQ;
    float d = 0.f;
    for (int n = 0; n < CFREQ; ++n) d += qi[n] * kj[n];
    sim[i][j] = d * ATT_SCALE;
  }
  __syncthreads();
  if (threadIdx.x < 32) {
    int i = threadIdx.x;
    float mx = -3.4e38f;
    for (int j = 0; j < 32; ++j) mx = fmaxf(mx, sim[i][j]);
    float s = 0.f;
    for (int j = 0; j < 32; ++j) { float e = expf(sim[i][j] - mx); sim[i][j] = e; s += e; }
    float inv = 1.f / s;
    for (int j = 0; j < 32; ++j) sim[i][j] *= inv;
  }
  __syncthreads();
  for (int e = threadIdx.x; e < 32 * CFREQ; e += TPB) {
    int i = e / CFREQ, n = e - i * CFREQ;
    float acc = 0.f;
    for (int j = 0; j < 32; ++j) acc += sim[i][j] * v[base + (size_t)j * CFREQ + n];
    out[base + (size_t)i * CFREQ + n] = acc;
  }
}

// ---------------------------------------------------------------------------
extern "C" void kernel_launch(void* const* d_in, const int* in_sizes, int n_in,
                              void* d_out, int out_size, void* d_ws, size_t ws_size,
                              hipStream_t stream) {
  const float* x    = (const float*)d_in[0];
  const float* gn_w = (const float*)d_in[1];
  const float* gn_b = (const float*)d_in[2];
  const float* dw_w = (const float*)d_in[3];
  const float* pw_w = (const float*)d_in[4];
  const float* pw_b = (const float*)d_in[5];
  const float* ao_w = (const float*)d_in[6];
  const float* ao_b = (const float*)d_in[7];
  const float* f1_w = (const float*)d_in[8];
  const float* f2_w = (const float*)d_in[9];
  const float* f3_w = (const float*)d_in[10];
  const float* f3_b = (const float*)d_in[11];
  float* out = (float*)d_out;
  char*  wsb = (char*)d_ws;

  const size_t SPAT4 = (size_t)CB * CC * CHW * 4;      // 128MiB (fp32 NCHW)
  const size_t SPAT2 = SPAT4 / 2;                      // bf16 version
  const size_t FRQ4  = (size_t)CB * CC * CFREQ * 4;    // 65MiB

  // --- liveness-based workspace (peak ~0.89 GiB) ---
  float*  xn     = (float*)(wsb + 0);                   // [1..8]
  float*  qkv    = (float*)(wsb + SPAT4);               // 3*SPAT4   [3..5]
  float*  ampB   = (float*)(wsb + 4 * SPAT4);           // 6*FRQ4    [5..6]
  float*  amp_q  = ampB;
  float*  ph_q   = (float*)((char*)ampB + 1 * FRQ4);
  float*  amp_k  = (float*)((char*)ampB + 2 * FRQ4);
  float*  ph_k   = (float*)((char*)ampB + 3 * FRQ4);
  float*  amp_v  = (float*)((char*)ampB + 4 * FRQ4);
  float*  ph_v   = (float*)((char*)ampB + 5 * FRQ4);
  __bf16* dwb16  = (__bf16*)(wsb + 4 * SPAT4);          // over amp region [2..3]
  float*  amp_o  = (float*)(wsb + SPAT4);               // over dead qkv   [6..7]
  float*  ph_o   = (float*)(wsb + SPAT4 + FRQ4);
  __bf16* outsp16= (__bf16*)(wsb + SPAT4 + 2 * FRQ4);   // [7..8]
  char*   attnOfs= wsb + SPAT4 + 2 * FRQ4 + SPAT2;
  float*  attn32 = (float*)attnOfs;                     // [8..11]
  __bf16* attn16 = (__bf16*)(attnOfs + SPAT4);          // [8..9]
  __bf16* h1b    = (__bf16*)(attnOfs + SPAT4 + SPAT2);  // 2*SPAT2   [9..10]
  __bf16* h2b    = (__bf16*)(attnOfs + SPAT4 + 3 * SPAT2); // [10..11]
  char*   wOfs   = wsb + 4 * SPAT4 + 6 * FRQ4;          // bf16 weights (5.4MB)
  __bf16* pwb    = (__bf16*)wOfs;                       // 768*256
  __bf16* aob    = pwb + 768 * 256;                     // 256*256
  __bf16* f1b    = aob + 256 * 256;                     // 512*2304 (tap-major)
  __bf16* f2b    = f1b + 512 * 2304;                    // 256*4608 (tap-major)
  __bf16* f3b    = f2b + 256 * 4608;                    // 256*256

  // 0. convert weights to bf16 once (3x3 weights permuted to tap-major K)
  k_cvt_bf16<<<(768 * 256 + TPB - 1) / TPB, TPB, 0, stream>>>(pw_w, pwb, 768 * 256);
  k_cvt_bf16<<<(256 * 256 + TPB - 1) / TPB, TPB, 0, stream>>>(ao_w, aob, 256 * 256);
  k_cvt_w3 <<<(512 * 2304 + TPB - 1) / TPB, TPB, 0, stream>>>(f1_w, f1b, 512, 256);
  k_cvt_w3 <<<(256 * 4608 + TPB - 1) / TPB, TPB, 0, stream>>>(f2_w, f2b, 256, 512);
  k_cvt_bf16<<<(256 * 256 + TPB - 1) / TPB, TPB, 0, stream>>>(f3_w, f3b, 256 * 256);

  // 1. GroupNorm
  k_groupnorm<<<CB * 32, TPB, 0, stream>>>(x, gn_w, gn_b, xn);
  // 2. depthwise 3x3 -> bf16
  k_dwconv3<<<(unsigned)((size_t)CB * CC * CHW / TPB), TPB, 0, stream>>>(xn, dw_w, dwb16);
  // 3. pointwise 1x1 -> qkv (fp32 for l2norm/FFT)
  k_conv_wmma<<<dim3(128, CQKV / 32, CB), TPB, 0, stream>>>(
      pwb, dwb16, pw_b, nullptr, qkv, nullptr, CQKV, CC, 1, 0);
  // 4. l2norm q,k,v (in-place)
  k_l2norm<<<512, TPB, 0, stream>>>(qkv, 0);
  k_l2norm<<<512, TPB, 0, stream>>>(qkv, 256);
  k_l2norm<<<512, TPB, 0, stream>>>(qkv, 512);
  // 5. rfft2 (ortho) -> amp/phase; one plane per workgroup, 128KB LDS
  k_rfft2_ap<<<CB * CC, TPB, 2 * CHW * sizeof(float), stream>>>(qkv, 0,   amp_q, ph_q);
  k_rfft2_ap<<<CB * CC, TPB, 2 * CHW * sizeof(float), stream>>>(qkv, 256, amp_k, ph_k);
  k_rfft2_ap<<<CB * CC, TPB, 2 * CHW * sizeof(float), stream>>>(qkv, 512, amp_v, ph_v);
  // 6. channel attention on amplitude and phase
  k_attn<<<CB * 8, TPB, 0, stream>>>(amp_q, amp_k, amp_v, amp_o);
  k_attn<<<CB * 8, TPB, 0, stream>>>(ph_q, ph_k, ph_v, ph_o);
  // 7. inverse rfft2 -> bf16
  k_irfft2<<<CB * CC, TPB, 2 * CHW * sizeof(float), stream>>>(amp_o, ph_o, outsp16);
  // 8. attn = ao(outsp) + xn (fp32 + bf16 outputs)
  k_conv_wmma<<<dim3(128, CC / 32, CB), TPB, 0, stream>>>(
      aob, outsp16, ao_b, xn, attn32, attn16, CC, CC, 1, 0);
  // 9. h1 = silu(conv3x3(attn, f1))  (tap-major implicit GEMM, K=2304)
  k_conv_wmma<<<dim3(128, 512 / 32, CB), TPB, 0, stream>>>(
      f1b, attn16, nullptr, nullptr, nullptr, h1b, 512, CC, 9, 1);
  // 10. h2 = conv3x3(h1, f2)         (tap-major implicit GEMM, K=4608)
  k_conv_wmma<<<dim3(128, CC / 32, CB), TPB, 0, stream>>>(
      f2b, h1b, nullptr, nullptr, nullptr, h2b, CC, 512, 9, 0);
  // 11. out = conv1x1(h2, f3) + f3_b + attn
  k_conv_wmma<<<dim3(128, CC / 32, CB), TPB, 0, stream>>>(
      f3b, h2b, f3_b, attn32, out, nullptr, CC, CC, 1, 0);
}
